// ProteinMotionMPNN_36000415875292
// MI455X (gfx1250) — compile-verified, matrix operand order checked by reading, not run on hardware
//
#include <hip/hip_runtime.h>
#include <hip/hip_bf16.h>

typedef _Float16 half_t;
typedef __attribute__((ext_vector_type(8)))  _Float16 v8h;
typedef __attribute__((ext_vector_type(16))) _Float16 v16h;
typedef __attribute__((ext_vector_type(8)))  float    v8f;

// ---------------- problem constants ----------------
#define NN      2048          // N nodes
#define EE      61440         // E edges
#define EMB_D   256
#define IN_D    1280
#define KNBR    30            // edges per node (row_index = e/30)
#define V3_D    15            // M*3
#define K1PAD   768           // MSG_IN (710) padded to 24*32
#define NGRAPH  8
#define LG      256           // nodes per graph
#define KCHUNK  64            // K-halves staged to LDS per chunk
#define BROW    72            // padded LDS row (halves): 144 B, 16B-aligned, bank-conflict-free

// ---------------- workspace layout (bytes) ----------------
static constexpr size_t OFF_XLNH  = 0;                         // 2048*1280 f16
static constexpr size_t OFF_EMBH  = OFF_XLNH  + 5242880;       // 2048*256 f16
static constexpr size_t OFF_INWT  = OFF_EMBH  + 1048576;       // 256*1280 f16
static constexpr size_t OFF_W1T   = OFF_INWT  + 655360;        // 3*256*768 f16
static constexpr size_t OFF_W2T   = OFF_W1T   + 1179648;       // 3*256*256 f16
static constexpr size_t OFF_W3T   = OFF_W2T   + 393216;        // 3*256*256 f16
static constexpr size_t OFF_TAIL  = OFF_W3T   + 393216;        // E*256 f16
static constexpr size_t OFF_H1    = OFF_TAIL  + 31457280;      // E*256 f16
static constexpr size_t OFF_H2    = OFF_H1    + 31457280;      // E*256 f16
static constexpr size_t OFF_MSG   = OFF_H2    + 31457280;      // E*256 f32
static constexpr size_t OFF_RB    = OFF_MSG   + 62914560;      // N*9 f32
static constexpr size_t OFF_TB    = OFF_RB    + 73728;         // N*3 f32
static constexpr size_t OFF_REDGE = OFF_TB    + 24576;         // E*9 f32
static constexpr size_t OFF_V     = OFF_REDGE + 2211840;       // N*15 f32
static constexpr size_t OFF_VTMP  = OFF_V     + 122880;        // N*15 f32
static constexpr size_t OFF_AGG   = OFF_VTMP  + 122880;        // N*256 f32

// ---------------- helpers ----------------
__device__ __forceinline__ float gelu_exact(float x) {
    return 0.5f * x * (1.0f + erff(x * 0.70710678118654752f));
}

__device__ __forceinline__ float block_reduce(float v, float* sh) {
    const int t = threadIdx.x;
    sh[t] = v;
    __syncthreads();
    #pragma unroll
    for (int s = 128; s > 0; s >>= 1) {
        if (t < s) sh[t] += sh[t + s];
        __syncthreads();
    }
    float r = sh[0];
    __syncthreads();
    return r;
}

__device__ __forceinline__ v16h pack16(v8h lo, v8h hi) {
    v16h o;
    #pragma unroll
    for (int i = 0; i < 8; ++i) { o[i] = lo[i]; o[8 + i] = hi[i]; }
    return o;
}

// A-fragment segment for the fused feat gather (layer MLP1).
// k (a multiple of 8 within a 32-block) never crosses the 256/512 boundaries.
__device__ __forceinline__ v8h feat_seg(int k, int gr, int gc, int e,
                                        const half_t* __restrict__ embh,
                                        const half_t* __restrict__ tail) {
    const half_t* p;
    if (k < 256)      p = embh + (size_t)gr * EMB_D + k;
    else if (k < 512) p = embh + (size_t)gc * EMB_D + (k - 256);
    else              p = tail + (size_t)e * 256 + (k - 512);
    return *(const v8h*)p;
}

// ---------------- WMMA GEMM: [rows x K] @ Bt^T -> [rows x 256] ----------------
// Block = 8 waves = 128 rows x 256 cols. Weights (Bt: f16 [256][Kpad], n-major)
// are staged to LDS in 256x64 chunks; each wave owns 16 rows and all 16
// 16-col output tiles (16 v8f accumulators), so A is fetched once per k-block.
__global__ void __launch_bounds__(256)
wmma_gemm_kernel(const half_t* __restrict__ A, int ldA,
                 const half_t* __restrict__ Bt, int Kpad, int kblocks,
                 const float* __restrict__ bias,
                 float* __restrict__ outF, half_t* __restrict__ outH,
                 int rows, int a_mode, int do_gelu,
                 const half_t* __restrict__ embh, const half_t* __restrict__ tail,
                 const int* __restrict__ rowi, const int* __restrict__ coli) {
    __shared__ half_t Bl[256 * BROW];                 // 36 KB
    const int w     = threadIdx.x >> 5;
    const int rtile = blockIdx.x * 8 + w;
    const int lane  = threadIdx.x & 31;
    const int r16   = lane & 15;
    const int ksub  = (lane >> 4) << 3;               // 0 or 8 (hw K pattern)
    const int arow  = rtile * 16 + r16;
    const bool act  = (rtile * 16 < rows);            // wave-uniform

    int gr = 0, gc = 0;
    if (act && a_mode) { gr = rowi[arow]; gc = coli[arow]; }

    v8f acc[16];
    #pragma unroll
    for (int t = 0; t < 16; ++t) acc[t] = (v8f){};

    const int nchunks = (kblocks * 32) / KCHUNK;
    for (int kc = 0; kc < nchunks; ++kc) {
        __syncthreads();
        // cooperative stage: 256 cols x 64 halves -> LDS (2048 16B segments)
        #pragma unroll
        for (int it = 0; it < 8; ++it) {
            const int s   = it * 256 + threadIdx.x;
            const int col = s >> 3, kseg = s & 7;
            v8h vseg = *(const v8h*)(Bt + (size_t)col * Kpad + kc * KCHUNK + kseg * 8);
            *(v8h*)&Bl[col * BROW + kseg * 8] = vseg;
        }
        __syncthreads();
        if (!act) continue;
        #pragma unroll
        for (int kbl = 0; kbl < 2; ++kbl) {
            const int kA = kc * KCHUNK + kbl * 32 + ksub;
            v16h afrag;
            if (a_mode == 0) {
                const half_t* p = A + (size_t)arow * ldA + kA;
                afrag = pack16(*(const v8h*)p, *(const v8h*)(p + 16));
            } else {
                afrag = pack16(feat_seg(kA,      gr, gc, arow, embh, tail),
                               feat_seg(kA + 16, gr, gc, arow, embh, tail));
            }
            const int kloc = kbl * 32 + ksub;
            #pragma unroll
            for (int t = 0; t < 16; ++t) {
                const half_t* bp = &Bl[(t * 16 + r16) * BROW + kloc];
                v16h bfrag = pack16(*(const v8h*)bp, *(const v8h*)(bp + 16));
                acc[t] = __builtin_amdgcn_wmma_f32_16x16x32_f16(
                    false, afrag, false, bfrag, (short)0, acc[t], false, false);
            }
        }
    }
    if (!act) return;

    const int mtop = (lane >> 4) << 3;                // D layout: M = mtop + vgpr
    #pragma unroll
    for (int t = 0; t < 16; ++t) {
        const int c0 = t * 16 + r16;
        const float bv = bias ? bias[c0] : 0.0f;
        #pragma unroll
        for (int r = 0; r < 8; ++r) {
            const int rr = rtile * 16 + mtop + r;
            float val = acc[t][r] + bv;
            if (do_gelu) val = gelu_exact(val);
            const size_t o = (size_t)rr * EMB_D + c0;
            if (outF) outF[o] = val;
            if (outH) outH[o] = (half_t)val;
        }
    }
}

// ---------------- small kernels ----------------
__global__ void zero_f32_kernel(float* p, int n) {
    int i = blockIdx.x * blockDim.x + threadIdx.x;
    if (i < n) p[i] = 0.0f;
}

// transpose f32 [K][Nn] -> f16 [Nn][Kpad] (zero-padded K)
__global__ void transpose_to_f16(const float* __restrict__ src, half_t* __restrict__ dst,
                                 int K, int Nn, int Kpad) {
    int idx = blockIdx.x * blockDim.x + threadIdx.x;
    if (idx >= Nn * Kpad) return;
    int n = idx / Kpad, k = idx % Kpad;
    float v = (k < K) ? src[(size_t)k * Nn + n] : 0.0f;
    dst[idx] = (half_t)v;
}

// LN over 1280 input features -> f16
__global__ void ln_input_kernel(const float* __restrict__ x, const float* __restrict__ g,
                                const float* __restrict__ b, half_t* __restrict__ out) {
    __shared__ float red[256];
    const int n = blockIdx.x, t = threadIdx.x;
    const float* xr = x + (size_t)n * IN_D;
    float vals[5]; float s = 0.0f;
    #pragma unroll
    for (int i = 0; i < 5; ++i) { vals[i] = xr[t + i * 256]; s += vals[i]; }
    const float mu = block_reduce(s, red) * (1.0f / IN_D);
    float vs = 0.0f;
    #pragma unroll
    for (int i = 0; i < 5; ++i) { float d = vals[i] - mu; vs += d * d; }
    const float inv = rsqrtf(block_reduce(vs, red) * (1.0f / IN_D) + 1e-5f);
    #pragma unroll
    for (int i = 0; i < 5; ++i) {
        const int c = t + i * 256;
        out[(size_t)n * IN_D + c] = (half_t)(g[c] * (vals[i] - mu) * inv + b[c]);
    }
}

// LN over 256 -> f32 (emb, in d_out) + f16 copy
__global__ void ln_emb_kernel(const float* __restrict__ h, const float* __restrict__ g,
                              const float* __restrict__ b, float* __restrict__ outF,
                              half_t* __restrict__ outH) {
    __shared__ float red[256];
    const int n = blockIdx.x, t = threadIdx.x;
    const float v  = h[(size_t)n * EMB_D + t];
    const float mu = block_reduce(v, red) * (1.0f / EMB_D);
    const float d  = v - mu;
    const float inv = rsqrtf(block_reduce(d * d, red) * (1.0f / EMB_D) + 1e-5f);
    const float o = g[t] * d * inv + b[t];
    outF[(size_t)n * EMB_D + t] = o;
    outH[(size_t)n * EMB_D + t] = (half_t)o;
}

__global__ void rigids_kernel(const float* __restrict__ bb, float* __restrict__ Rb,
                              float* __restrict__ Tb) {
    int n = blockIdx.x * blockDim.x + threadIdx.x;
    if (n >= NN) return;
    const float* p = bb + (size_t)n * 9;
    float nx = p[0], ny = p[1], nz = p[2];
    float cax = p[3], cay = p[4], caz = p[5];
    float cx = p[6], cy = p[7], cz = p[8];
    float e1x = cx - cax, e1y = cy - cay, e1z = cz - caz;
    float inv = rsqrtf(e1x * e1x + e1y * e1y + e1z * e1z + 1e-30f);
    e1x *= inv; e1y *= inv; e1z *= inv;
    float ux = nx - cax, uy = ny - cay, uz = nz - caz;
    const float du = ux * e1x + uy * e1y + uz * e1z;
    float e2x = ux - du * e1x, e2y = uy - du * e1y, e2z = uz - du * e1z;
    inv = rsqrtf(e2x * e2x + e2y * e2y + e2z * e2z + 1e-30f);
    e2x *= inv; e2y *= inv; e2z *= inv;
    const float e3x = e1y * e2z - e1z * e2y;
    const float e3y = e1z * e2x - e1x * e2z;
    const float e3z = e1x * e2y - e1y * e2x;
    float* R = Rb + (size_t)n * 9;           // columns are e1,e2,e3
    R[0] = e1x; R[1] = e2x; R[2] = e3x;
    R[3] = e1y; R[4] = e2y; R[5] = e3y;
    R[6] = e1z; R[7] = e2z; R[8] = e3z;
    Tb[(size_t)n * 3 + 0] = cax; Tb[(size_t)n * 3 + 1] = cay; Tb[(size_t)n * 3 + 2] = caz;
}

// Edge features: R_edge, quats (power iteration), t_edge, chain/log dist, radial.
// Writes tail[e][45..255] (edge_attr + zero pad); tail[0..44] filled per-layer.
__global__ void edge_kernel(const float* __restrict__ bb, const float* __restrict__ Rb,
                            const float* __restrict__ Tb, const int* __restrict__ rowi,
                            const int* __restrict__ coli, float* __restrict__ Redge,
                            half_t* __restrict__ tail) {
    int e = blockIdx.x * blockDim.x + threadIdx.x;
    if (e >= EE) return;
    const int i = rowi[e], j = coli[e];
    float Ri[9], Rj[9], Re[9];
    #pragma unroll
    for (int k = 0; k < 9; ++k) { Ri[k] = Rb[(size_t)i * 9 + k]; Rj[k] = Rb[(size_t)j * 9 + k]; }
    #pragma unroll
    for (int a = 0; a < 3; ++a)
        #pragma unroll
        for (int c = 0; c < 3; ++c)
            Re[a * 3 + c] = Ri[0 + a] * Rj[0 + c] + Ri[3 + a] * Rj[3 + c] + Ri[6 + a] * Rj[6 + c];
    #pragma unroll
    for (int k = 0; k < 9; ++k) Redge[(size_t)e * 9 + k] = Re[k];

    const float dt0 = Tb[(size_t)j * 3 + 0] - Tb[(size_t)i * 3 + 0];
    const float dt1 = Tb[(size_t)j * 3 + 1] - Tb[(size_t)i * 3 + 1];
    const float dt2 = Tb[(size_t)j * 3 + 2] - Tb[(size_t)i * 3 + 2];
    float te[3];
    #pragma unroll
    for (int a = 0; a < 3; ++a) te[a] = Ri[0 + a] * dt0 + Ri[3 + a] * dt1 + Ri[6 + a] * dt2;

    // quaternion via shifted power iteration on Davenport K/3
    const float xx = Re[0], xy = Re[1], xz = Re[2];
    const float yx = Re[3], yy = Re[4], yz = Re[5];
    const float zx = Re[6], zy = Re[7], zz = Re[8];
    float Kq[4][4] = {
        { xx + yy + zz, zy - yz,      xz - zx,      yx - xy      },
        { zy - yz,      xx - yy - zz, xy + yx,      xz + zx      },
        { xz - zx,      xy + yx,      yy - xx - zz, yz + zy      },
        { yx - xy,      xz + zx,      yz + zy,      zz - xx - yy }};
    #pragma unroll
    for (int a = 0; a < 4; ++a)
        #pragma unroll
        for (int b2 = 0; b2 < 4; ++b2) Kq[a][b2] *= (1.0f / 3.0f);
    float q[4] = {1.0f, 0.0f, 0.0f, 0.0f};
    for (int it = 0; it < 32; ++it) {
        float nq[4];
        #pragma unroll
        for (int a = 0; a < 4; ++a)
            nq[a] = q[a] + Kq[a][0] * q[0] + Kq[a][1] * q[1] + Kq[a][2] * q[2] + Kq[a][3] * q[3];
        const float nn = rsqrtf(nq[0] * nq[0] + nq[1] * nq[1] + nq[2] * nq[2] + nq[3] * nq[3] + 1e-30f);
        #pragma unroll
        for (int a = 0; a < 4; ++a) q[a] = nq[a] * nn;
    }

    half_t* tp = tail + (size_t)e * 256;
    tp[45] = (half_t)q[0]; tp[46] = (half_t)q[1]; tp[47] = (half_t)q[2]; tp[48] = (half_t)q[3];
    tp[49] = (half_t)te[0]; tp[50] = (half_t)te[1]; tp[51] = (half_t)te[2];
    tp[52] = (half_t)logf(fabsf((float)(i - j)) + 1.0f);
    tp[53] = (half_t)logf(sqrtf(te[0] * te[0] + te[1] * te[1] + te[2] * te[2]) + 1e-8f);
    #pragma unroll
    for (int a = 0; a < 3; ++a) {
        #pragma unroll
        for (int b2 = 0; b2 < 3; ++b2) {
            const float dx = bb[(size_t)i * 9 + a * 3 + 0] - bb[(size_t)j * 9 + b2 * 3 + 0];
            const float dy = bb[(size_t)i * 9 + a * 3 + 1] - bb[(size_t)j * 9 + b2 * 3 + 1];
            const float dz = bb[(size_t)i * 9 + a * 3 + 2] - bb[(size_t)j * 9 + b2 * 3 + 2];
            const float d = sqrtf(fmaxf(dx * dx + dy * dy + dz * dz, 1e-12f));
            for (int s = 0; s < 16; ++s) {
                const float mu = (20.0f / 15.0f) * (float)s;
                const float dd = d - mu;
                tp[54 + (a * 3 + b2) * 16 + s] = (half_t)expf(-0.5f * dd * dd);
            }
        }
    }
    for (int k = 198; k < 256; ++k) tp[k] = (half_t)0.0f;   // pad k 710..767
}

// per-layer: tail[e][0..44] = [v_i, R_edge@v_j, v_i - R_edge@v_j]
__global__ void tail_update_kernel(const float* __restrict__ v, const float* __restrict__ Redge,
                                   const int* __restrict__ rowi, const int* __restrict__ coli,
                                   half_t* __restrict__ tail) {
    int e = blockIdx.x * blockDim.x + threadIdx.x;
    if (e >= EE) return;
    const int i = rowi[e], j = coli[e];
    float Re[9];
    #pragma unroll
    for (int k = 0; k < 9; ++k) Re[k] = Redge[(size_t)e * 9 + k];
    half_t* tp = tail + (size_t)e * 256;
    #pragma unroll
    for (int m = 0; m < 5; ++m) {
        const float vj0 = v[(size_t)j * V3_D + m * 3 + 0];
        const float vj1 = v[(size_t)j * V3_D + m * 3 + 1];
        const float vj2 = v[(size_t)j * V3_D + m * 3 + 2];
        #pragma unroll
        for (int a = 0; a < 3; ++a) {
            const float vi = v[(size_t)i * V3_D + m * 3 + a];
            const float vr = Re[a * 3 + 0] * vj0 + Re[a * 3 + 1] * vj1 + Re[a * 3 + 2] * vj2;
            tp[m * 3 + a]      = (half_t)vi;
            tp[15 + m * 3 + a] = (half_t)vr;
            tp[30 + m * 3 + a] = (half_t)(vi - vr);
        }
    }
}

// agg[n][c] = emb[n][c] + mean over 30 consecutive edges of msg
__global__ void agg_kernel(const float* __restrict__ msg, const float* __restrict__ emb,
                           float* __restrict__ agg) {
    const int idx = blockIdx.x * blockDim.x + threadIdx.x;   // n*256 + c
    const int n = idx >> 8, c = idx & 255;
    const float* mp = msg + ((size_t)n * KNBR) * EMB_D + c;
    float s = 0.0f;
    #pragma unroll
    for (int e = 0; e < KNBR; ++e) s += mp[(size_t)e * EMB_D];
    agg[idx] = emb[idx] + s * (1.0f / KNBR);
}

// v_tmp = v + [emb, v] @ vect_W[l]  (271 x 15)
__global__ void vupdate_kernel(const float* __restrict__ emb, const float* __restrict__ v,
                               const float* __restrict__ W, float* __restrict__ vtmp) {
    const int idx = blockIdx.x * blockDim.x + threadIdx.x;   // n*15 + o
    if (idx >= NN * V3_D) return;
    const int n = idx / V3_D, o = idx % V3_D;
    float s = v[idx];
    const float* er = emb + (size_t)n * EMB_D;
    for (int k = 0; k < EMB_D; ++k) s += er[k] * W[k * V3_D + o];
    const float* vr = v + (size_t)n * V3_D;
    #pragma unroll
    for (int k = 0; k < V3_D; ++k) s += vr[k] * W[(EMB_D + k) * V3_D + o];
    vtmp[idx] = s;
}

// center over L per (m,xyz), then MGS-QR of the 768x5 view, scale by sqrt(L)=16
__global__ void qr_kernel(const float* __restrict__ vin, float* __restrict__ vout) {
    __shared__ float mat[LG * V3_D];   // 3840
    __shared__ float red[256];
    const int g = blockIdx.x, t = threadIdx.x;
    const float* src = vin + (size_t)g * (LG * V3_D);
    for (int i = t; i < LG * V3_D; i += 256) mat[i] = src[i];
    __syncthreads();
    // center each of 15 channels over the 256 rows
    float cv[V3_D];
    #pragma unroll
    for (int c = 0; c < V3_D; ++c) cv[c] = mat[t * V3_D + c];
    for (int c = 0; c < V3_D; ++c) {
        const float mu = block_reduce(cv[c], red) * (1.0f / LG);
        mat[t * V3_D + c] = cv[c] - mu;
    }
    __syncthreads();
    // MGS on the 768x5 reinterpretation; thread owns rows t, t+256, t+512
    for (int c = 0; c < 5; ++c) {
        for (int p = 0; p < c; ++p) {
            float d = 0.0f;
            for (int r = t; r < 768; r += 256) d += mat[r * 5 + c] * mat[r * 5 + p];
            const float dot = block_reduce(d, red);
            for (int r = t; r < 768; r += 256) mat[r * 5 + c] -= dot * mat[r * 5 + p];
            __syncthreads();
        }
        float s2 = 0.0f;
        for (int r = t; r < 768; r += 256) { const float vv = mat[r * 5 + c]; s2 += vv * vv; }
        const float nrm = sqrtf(fmaxf(block_reduce(s2, red), 1e-24f));
        const float scale = 16.0f / nrm;
        for (int r = t; r < 768; r += 256) mat[r * 5 + c] *= scale;
        __syncthreads();
    }
    for (int i = t; i < LG * V3_D; i += 256) vout[(size_t)g * (LG * V3_D) + i] = mat[i];
}

// v_global[n][m] = R[n] @ v[n][m]
__global__ void vglobal_kernel(const float* __restrict__ Rb, const float* __restrict__ v,
                               float* __restrict__ out) {
    const int idx = blockIdx.x * blockDim.x + threadIdx.x;   // n*5 + m
    if (idx >= NN * 5) return;
    const int n = idx / 5, m = idx % 5;
    const float x = v[(size_t)n * V3_D + m * 3 + 0];
    const float y = v[(size_t)n * V3_D + m * 3 + 1];
    const float z = v[(size_t)n * V3_D + m * 3 + 2];
    const float* R = Rb + (size_t)n * 9;
    out[(size_t)n * V3_D + m * 3 + 0] = R[0] * x + R[1] * y + R[2] * z;
    out[(size_t)n * V3_D + m * 3 + 1] = R[3] * x + R[4] * y + R[5] * z;
    out[(size_t)n * V3_D + m * 3 + 2] = R[6] * x + R[7] * y + R[8] * z;
}

// ---------------- launch ----------------
extern "C" void kernel_launch(void* const* d_in, const int* in_sizes, int n_in,
                              void* d_out, int out_size, void* d_ws, size_t ws_size,
                              hipStream_t stream) {
    const float* x       = (const float*)d_in[0];
    const float* bb      = (const float*)d_in[1];
    const float* in_ln_g = (const float*)d_in[2];
    const float* in_ln_b = (const float*)d_in[3];
    const float* in_W    = (const float*)d_in[4];
    const float* in_b    = (const float*)d_in[5];
    const float* msg_W1  = (const float*)d_in[6];
    const float* msg_b1  = (const float*)d_in[7];
    const float* msg_W2  = (const float*)d_in[8];
    const float* msg_b2  = (const float*)d_in[9];
    const float* msg_W3  = (const float*)d_in[10];
    const float* msg_b3  = (const float*)d_in[11];
    const float* vect_W  = (const float*)d_in[12];
    const float* ln_g    = (const float*)d_in[13];
    const float* ln_b    = (const float*)d_in[14];
    const int*   rowi    = (const int*)d_in[15];
    const int*   coli    = (const int*)d_in[16];
    (void)in_sizes; (void)n_in; (void)out_size; (void)ws_size;

    char* ws = (char*)d_ws;
    half_t* xlnh = (half_t*)(ws + OFF_XLNH);
    half_t* embh = (half_t*)(ws + OFF_EMBH);
    half_t* inWt = (half_t*)(ws + OFF_INWT);
    half_t* W1t  = (half_t*)(ws + OFF_W1T);
    half_t* W2t  = (half_t*)(ws + OFF_W2T);
    half_t* W3t  = (half_t*)(ws + OFF_W3T);
    half_t* tail = (half_t*)(ws + OFF_TAIL);
    half_t* h1   = (half_t*)(ws + OFF_H1);
    half_t* h2   = (half_t*)(ws + OFF_H2);
    float*  msg  = (float*)(ws + OFF_MSG);
    float*  Rb   = (float*)(ws + OFF_RB);
    float*  Tb   = (float*)(ws + OFF_TB);
    float*  Redg = (float*)(ws + OFF_REDGE);
    float*  vbuf = (float*)(ws + OFF_V);
    float*  vtmp = (float*)(ws + OFF_VTMP);
    float*  aggb = (float*)(ws + OFF_AGG);
    float*  emb  = (float*)d_out;                         // [N][256] (final output slot)
    float*  vglb = (float*)d_out + (size_t)NN * EMB_D;    // [N][15]

    // input LN -> f16, weight transposes -> f16 B-layouts
    ln_input_kernel<<<NN, 256, 0, stream>>>(x, in_ln_g, in_ln_b, xlnh);
    transpose_to_f16<<<(256 * 1280) / 256, 256, 0, stream>>>(in_W, inWt, 1280, 256, 1280);
    for (int l = 0; l < 3; ++l) {
        transpose_to_f16<<<(256 * K1PAD) / 256, 256, 0, stream>>>(
            msg_W1 + (size_t)l * 710 * 256, W1t + (size_t)l * 256 * K1PAD, 710, 256, K1PAD);
        transpose_to_f16<<<(256 * 256) / 256, 256, 0, stream>>>(
            msg_W2 + (size_t)l * 256 * 256, W2t + (size_t)l * 256 * 256, 256, 256, 256);
        transpose_to_f16<<<(256 * 256) / 256, 256, 0, stream>>>(
            msg_W3 + (size_t)l * 256 * 256, W3t + (size_t)l * 256 * 256, 256, 256, 256);
    }

    // emb = LN(x) @ in_W + in_b   (WMMA, K=1280; block = 128 rows)
    wmma_gemm_kernel<<<NN / 128, 256, 0, stream>>>(
        xlnh, IN_D, inWt, IN_D, IN_D / 32, in_b, emb, embh, NN, 0, 0,
        nullptr, nullptr, nullptr, nullptr);

    // geometry
    rigids_kernel<<<NN / 256, 256, 0, stream>>>(bb, Rb, Tb);
    zero_f32_kernel<<<(NN * V3_D) / 256, 256, 0, stream>>>(vbuf, NN * V3_D);
    edge_kernel<<<EE / 256, 256, 0, stream>>>(bb, Rb, Tb, rowi, coli, Redg, tail);

    const int mlp_blocks = EE / 128;   // 480
    for (int l = 0; l < 3; ++l) {
        tail_update_kernel<<<EE / 256, 256, 0, stream>>>(vbuf, Redg, rowi, coli, tail);
        // h1 = gelu(feat @ W1 + b1) — fused gather A (K=768)
        wmma_gemm_kernel<<<mlp_blocks, 256, 0, stream>>>(
            nullptr, 0, W1t + (size_t)l * 256 * K1PAD, K1PAD, K1PAD / 32,
            msg_b1 + (size_t)l * 256, nullptr, h1, EE, 1, 1, embh, tail, rowi, coli);
        // h2 = gelu(h1 @ W2 + b2)
        wmma_gemm_kernel<<<mlp_blocks, 256, 0, stream>>>(
            h1, 256, W2t + (size_t)l * 256 * 256, 256, 8,
            msg_b2 + (size_t)l * 256, nullptr, h2, EE, 0, 1, nullptr, nullptr, nullptr, nullptr);
        // msg = h2 @ W3 + b3  (f32)
        wmma_gemm_kernel<<<mlp_blocks, 256, 0, stream>>>(
            h2, 256, W3t + (size_t)l * 256 * 256, 256, 8,
            msg_b3 + (size_t)l * 256, msg, nullptr, EE, 0, 0, nullptr, nullptr, nullptr, nullptr);
        // contiguous segment-mean + residual, then LN
        agg_kernel<<<NN, 256, 0, stream>>>(msg, emb, aggb);
        ln_emb_kernel<<<NN, 256, 0, stream>>>(aggb, ln_g + (size_t)l * 256, ln_b + (size_t)l * 256,
                                              emb, embh);
        // vector channel update + per-graph QR normalization
        vupdate_kernel<<<(NN * V3_D) / 256, 256, 0, stream>>>(
            emb, vbuf, vect_W + (size_t)l * 271 * V3_D, vtmp);
        qr_kernel<<<NGRAPH, 256, 0, stream>>>(vtmp, vbuf);
    }

    vglobal_kernel<<<(NN * 5) / 256, 256, 0, stream>>>(Rb, vbuf, vglb);
}
